// SimplifiedRetention_23974507446365
// MI455X (gfx1250) — compile-verified, matrix operand order checked by reading, not run on hardware
//
#include <hip/hip_runtime.h>

// ---------------------------------------------------------------------------
// Types for CDNA5 WMMA (wave32): v_wmma_f32_16x16x32_bf16
//   A: 16x32 bf16 (v16bf per lane), B: 32x16 bf16 (v16bf), C/D: 16x16 f32 (v8f)
// ---------------------------------------------------------------------------
typedef __bf16 bf16_t;
typedef __attribute__((ext_vector_type(16))) __bf16 v16bf;
typedef __attribute__((ext_vector_type(8)))  __bf16 v8bf;
typedef __attribute__((ext_vector_type(8)))  float  v8f;

static __device__ __forceinline__ bf16_t f2bf(float f) {
  // round-to-nearest-even f32 -> bf16
  unsigned u = __builtin_bit_cast(unsigned, f);
  unsigned r = u + 0x7fffu + ((u >> 16) & 1u);
  unsigned short s = (unsigned short)(r >> 16);
  return __builtin_bit_cast(bf16_t, s);
}

// Per-lane 16-element fragment: elements 0..7 contiguous at p, 8..15 at p+16.
// Combined with koff = (lane<16 ? 0 : 8) this matches the ISA 16-bit A/B
// layout: lo lanes hold K 0..7,16..23 ; hi lanes hold K 8..15,24..31.
static __device__ __forceinline__ v16bf load16(const bf16_t* p) {
  union { v16bf v; v8bf h[2]; } u;
  u.h[0] = *(const v8bf*)(p);
  u.h[1] = *(const v8bf*)(p + 16);
  return u.v;
}

static __device__ __forceinline__ v8f wmma_bf16(v16bf a, v16bf b, v8f c) {
  return __builtin_amdgcn_wmma_f32_16x16x32_bf16(
      /*neg_a=*/false, a, /*neg_b=*/false, b,
      /*c_mod=*/(short)0, c, /*reuse_a=*/false, /*reuse_b=*/false);
}

// ---------------------------------------------------------------------------
// f32 -> bf16 elementwise convert
// ---------------------------------------------------------------------------
__global__ void cvt_bf16_kernel(const float* __restrict__ in,
                                bf16_t* __restrict__ out, int n) {
  int i = blockIdx.x * blockDim.x + threadIdx.x;
  if (i < n) out[i] = f2bf(in[i]);
}

// ---------------------------------------------------------------------------
// GEMM: C[M,N] (f32) = A[M,K] (bf16, row-major) @ Bt[N,K]^T (bf16, row-major)
// i.e. y = x @ W^T with W stored [N,K]. Both operands K-contiguous -> every
// WMMA fragment is a contiguous 16B global_load_b128.
// Block 256 thr = 8 waves (4 along M x 2 along N); block tile 128x128;
// wave tile 32Mx64N = 8 WMMAs per K=32 step on 2 A + 4 B fragments
// (~21 cached FLOP/byte, vs 10.7 for a 16x32 wave tile).
// ---------------------------------------------------------------------------
__global__ void gemm_bf16_wmma_kernel(const bf16_t* __restrict__ A,
                                      const bf16_t* __restrict__ Bt,
                                      float* __restrict__ C,
                                      int M, int N, int K) {
  const int lane = threadIdx.x & 31;
  const int wid  = threadIdx.x >> 5;
  const int r    = lane & 15;
  const int hi   = lane >> 4;     // 0 | 1
  const int koff = hi * 8;

  const int mBase = blockIdx.y * 128 + (wid & 3) * 32;   // 4 waves along M
  const int nBase = blockIdx.x * 128 + (wid >> 2) * 64;  // 2 waves along N

  const bf16_t* aRow0 = A  + (long)(mBase + r)      * K + koff;
  const bf16_t* aRow1 = A  + (long)(mBase + 16 + r) * K + koff;
  const bf16_t* bRow0 = Bt + (long)(nBase + r)      * K + koff;
  const bf16_t* bRow1 = Bt + (long)(nBase + 16 + r) * K + koff;
  const bf16_t* bRow2 = Bt + (long)(nBase + 32 + r) * K + koff;
  const bf16_t* bRow3 = Bt + (long)(nBase + 48 + r) * K + koff;

  v8f acc[2][4] = {};

#pragma unroll 2
  for (int k = 0; k < K; k += 32) {
    __builtin_prefetch(aRow0 + k + 256, 0, 1);   // global_prefetch_b8
    v16bf a0 = load16(aRow0 + k);
    v16bf a1 = load16(aRow1 + k);
    v16bf b0 = load16(bRow0 + k);
    v16bf b1 = load16(bRow1 + k);
    v16bf b2 = load16(bRow2 + k);
    v16bf b3 = load16(bRow3 + k);
    acc[0][0] = wmma_bf16(a0, b0, acc[0][0]);
    acc[0][1] = wmma_bf16(a0, b1, acc[0][1]);
    acc[0][2] = wmma_bf16(a0, b2, acc[0][2]);
    acc[0][3] = wmma_bf16(a0, b3, acc[0][3]);
    acc[1][0] = wmma_bf16(a1, b0, acc[1][0]);
    acc[1][1] = wmma_bf16(a1, b1, acc[1][1]);
    acc[1][2] = wmma_bf16(a1, b2, acc[1][2]);
    acc[1][3] = wmma_bf16(a1, b3, acc[1][3]);
  }

#pragma unroll
  for (int ti = 0; ti < 2; ++ti) {
#pragma unroll
    for (int tj = 0; tj < 4; ++tj) {
      union { v8f v; float f[8]; } u;
      u.v = acc[ti][tj];
#pragma unroll
      for (int i = 0; i < 8; ++i) {
        int Mm = ti * 16 + i + hi * 8;   // C/D layout: M = vgpr + 8*laneHi
        C[(long)(mBase + Mm) * N + nBase + tj * 16 + r] = u.f[i];
      }
    }
  }
}

// ---------------------------------------------------------------------------
// RoPE (on f32 [B,S,H,Dh]) fused with bf16 convert, same layout out.
// One thread per (b,s,h,d) with d < Dh/2 (handles the rotated pair).
// ---------------------------------------------------------------------------
__global__ void rope_bf16_kernel(const float* __restrict__ in,
                                 bf16_t* __restrict__ out,
                                 int B, int S, int H, int Dh) {
  const int half = Dh / 2;
  int idx = blockIdx.x * blockDim.x + threadIdx.x;
  int total = B * S * H * half;
  if (idx >= total) return;
  int d2 = idx % half; int t = idx / half;
  int h  = t % H;      t /= H;
  int s  = t % S;      int b = t / S;

  float inv_freq = __builtin_exp2f(-((2.0f * (float)d2) / (float)Dh) * 13.2877123795f); // log2(10000)
  float theta = (float)s * inv_freq;
  float sn = __builtin_sinf(theta);
  float cs = __builtin_cosf(theta);

  long base = (((long)b * S + s) * H + h) * Dh;
  float x1 = in[base + d2];
  float x2 = in[base + d2 + half];
  out[base + d2]        = f2bf(x1 * cs - x2 * sn);
  out[base + d2 + half] = f2bf(x2 * cs + x1 * sn);
}

// ---------------------------------------------------------------------------
// V: f32 [B,S,H,Dh] -> bf16 transposed [B,H,Dh,S] so the PV-WMMA B-matrix
// fragments (runs over key index) are contiguous.
// ---------------------------------------------------------------------------
__global__ void vT_bf16_kernel(const float* __restrict__ in,
                               bf16_t* __restrict__ out,
                               int B, int S, int H, int Dh) {
  int idx = blockIdx.x * blockDim.x + threadIdx.x;
  int total = B * S * H * Dh;
  if (idx >= total) return;
  int d = idx % Dh; int t = idx / Dh;
  int h = t % H;    t /= H;
  int s = t % S;    int b = t / S;
  out[(((long)b * H + h) * Dh + d) * S + s] = f2bf(in[idx]);
}

// ---------------------------------------------------------------------------
// Streaming retention (flash-style, no softmax -> pure linear accumulation):
//   out[s,:] = sum_{t<=s} (q_s . k_t / sqrt(Dh)) * gamma_h^(s-t) * v[t,:]
// Block = 8 waves, each wave owns 16 query rows; loop key chunks of 32.
// Per chunk: 4 WMMAs QK^T (Dh=64), decay+mask on f32 score regs, bf16 LDS
// round-trip (C-layout -> A-layout transpose), 4 WMMAs S@V.
// Avoids materializing the 512MB [B,H,S,S] retention matrix entirely.
// ---------------------------------------------------------------------------
__global__ void retention_kernel(const bf16_t* __restrict__ qb,   // [B,S,H,Dh]
                                 const bf16_t* __restrict__ kb,   // [B,S,H,Dh]
                                 const bf16_t* __restrict__ vt,   // [B,H,Dh,S]
                                 float* __restrict__ att,         // [B,S,H,Dh]
                                 int B, int S, int H, int Dh) {
  __shared__ bf16_t smem[8 * 16 * 32];                 // 1KB per wave
  const int lane = threadIdx.x & 31;
  const int wid  = threadIdx.x >> 5;
  const int r    = lane & 15;
  const int hi   = lane >> 4;
  const int koff = hi * 8;

  const int bh = blockIdx.x;
  const int b  = bh / H;
  const int h  = bh % H;
  const int qBase = blockIdx.y * 128 + wid * 16;

  const float gamma = 1.0f - __builtin_exp2f(-5.0f - (float)h);
  const float lg    = __builtin_log2f(gamma);
  const float scale = 0.125f;                          // 1/sqrt(64)

  // Preload Q A-tiles for this wave's 16 rows (Dh=64 -> two K=32 tiles)
  const bf16_t* qRow = qb + (((long)b * S + qBase + r) * H + h) * Dh + koff;
  v16bf qa0 = load16(qRow);
  v16bf qa1 = load16(qRow + 32);

  v8f o0 = {}, o1 = {}, o2 = {}, o3 = {};
  bf16_t* sws = smem + wid * 512;

  for (int tBase = 0; tBase < qBase + 16; tBase += 32) {
    // ---- scores = Q @ K^T over Dh (two 16x16 tiles along keys) ----
    const bf16_t* kRow0 = kb + (((long)b * S + tBase + r)      * H + h) * Dh + koff;
    const bf16_t* kRow1 = kb + (((long)b * S + tBase + 16 + r) * H + h) * Dh + koff;
    v16bf kb00 = load16(kRow0);
    v16bf kb01 = load16(kRow0 + 32);
    v16bf kb10 = load16(kRow1);
    v16bf kb11 = load16(kRow1 + 32);

    v8f s0 = {}, s1 = {};
    s0 = wmma_bf16(qa0, kb00, s0);
    s0 = wmma_bf16(qa1, kb01, s0);
    s1 = wmma_bf16(qa0, kb10, s1);
    s1 = wmma_bf16(qa1, kb11, s1);

    // ---- decay * causal mask, write bf16 scores to LDS (row-major 16x32) ---
    union { v8f v; float f[8]; } u0, u1;
    u0.v = s0; u1.v = s1;
#pragma unroll
    for (int i = 0; i < 8; ++i) {
      int Mm = i + hi * 8;
      int sIdx = qBase + Mm;
      int d0 = sIdx - (tBase + r);         // tile0: key col = tBase + r
      int d1 = d0 - 16;                    // tile1: key col = tBase + 16 + r
      float e0 = (d0 >= 0) ? scale * __builtin_exp2f((float)d0 * lg) * u0.f[i] : 0.0f;
      float e1 = (d1 >= 0) ? scale * __builtin_exp2f((float)d1 * lg) * u1.f[i] : 0.0f;
      sws[Mm * 32 + r]      = f2bf(e0);
      sws[Mm * 32 + 16 + r] = f2bf(e1);
    }
    // Read back in A layout (same-wave LDS ops are in-order; compiler
    // inserts s_wait_dscnt for the RAW dependency).
    v16bf sa = load16(sws + r * 32 + koff);

    // ---- out += S @ V  (4 output tiles across Dh) ----
    const bf16_t* vBase = vt + ((long)bh * Dh + r) * S + tBase + koff;
    v16bf vb0 = load16(vBase);
    v16bf vb1 = load16(vBase + 16L * S);
    v16bf vb2 = load16(vBase + 32L * S);
    v16bf vb3 = load16(vBase + 48L * S);
    o0 = wmma_bf16(sa, vb0, o0);
    o1 = wmma_bf16(sa, vb1, o1);
    o2 = wmma_bf16(sa, vb2, o2);
    o3 = wmma_bf16(sa, vb3, o3);
  }

  union { v8f v; float f[8]; } a0, a1, a2, a3;
  a0.v = o0; a1.v = o1; a2.v = o2; a3.v = o3;
#pragma unroll
  for (int i = 0; i < 8; ++i) {
    int Mm = i + hi * 8;
    long row = (((long)b * S + qBase + Mm) * H + h) * Dh;
    att[row + r]      = a0.f[i];
    att[row + 16 + r] = a1.f[i];
    att[row + 32 + r] = a2.f[i];
    att[row + 48 + r] = a3.f[i];
  }
}

// ---------------------------------------------------------------------------
// gated[i] = bf16( att[i] * silu(g[i]) )
// ---------------------------------------------------------------------------
__global__ void gate_silu_kernel(const float* __restrict__ att,
                                 const float* __restrict__ g,
                                 bf16_t* __restrict__ out, int n) {
  int i = blockIdx.x * blockDim.x + threadIdx.x;
  if (i >= n) return;
  float gv = g[i];
  float s  = gv / (1.0f + __builtin_expf(-gv));
  out[i] = f2bf(att[i] * s);
}

// ---------------------------------------------------------------------------
// Host-side orchestration
// ---------------------------------------------------------------------------
extern "C" void kernel_launch(void* const* d_in, const int* in_sizes, int n_in,
                              void* d_out, int out_size, void* d_ws, size_t ws_size,
                              hipStream_t stream) {
  const int S  = 2048, H = 16;
  const int D  = 1024, Dh = D / H;
  const int MB = in_sizes[0] / D;      // B*S = 4096
  const int B  = MB / S;               // 2

  const float* x  = (const float*)d_in[0];
  const float* wq = (const float*)d_in[1];
  const float* wk = (const float*)d_in[2];
  const float* wv = (const float*)d_in[3];
  const float* wg = (const float*)d_in[4];
  const float* wo = (const float*)d_in[5];
  float* out = (float*)d_out;

  // workspace layout
  char* ws = (char*)d_ws;
  const size_t szXb  = (size_t)MB * D * sizeof(bf16_t);   // 8 MB
  const size_t szWb  = (size_t)D  * D * sizeof(bf16_t);   // 2 MB each
  const size_t szF32 = (size_t)MB * D * sizeof(float);    // 16 MB

  bf16_t* xb    = (bf16_t*)(ws);                 size_t off = szXb;
  bf16_t* wqb   = (bf16_t*)(ws + off);           off += szWb;
  bf16_t* wkb   = (bf16_t*)(ws + off);           off += szWb;
  bf16_t* wvb   = (bf16_t*)(ws + off);           off += szWb;
  bf16_t* wgb   = (bf16_t*)(ws + off);           off += szWb;
  bf16_t* wob   = (bf16_t*)(ws + off);           off += szWb;
  float*  qf    = (float*) (ws + off);           off += szF32;
  float*  kf    = (float*) (ws + off);           off += szF32;
  float*  vf    = (float*) (ws + off);           off += szF32;
  float*  gf    = (float*) (ws + off);           off += szF32;
  bf16_t* qbb   = (bf16_t*)(ws + off);           off += szXb;
  bf16_t* kbb   = (bf16_t*)(ws + off);           off += szXb;
  bf16_t* vtb   = (bf16_t*)(ws + off);           off += szXb;
  float*  attf  = (float*) (ws + off);           off += szF32;
  bf16_t* gatedb= (bf16_t*)(ws + off);           off += szXb;

  const int TPB = 256;
  auto blocks = [&](int n) { return (n + TPB - 1) / TPB; };

  // 1) bf16 conversions
  cvt_bf16_kernel<<<blocks(MB * D), TPB, 0, stream>>>(x,  xb,  MB * D);
  cvt_bf16_kernel<<<blocks(D * D),  TPB, 0, stream>>>(wq, wqb, D * D);
  cvt_bf16_kernel<<<blocks(D * D),  TPB, 0, stream>>>(wk, wkb, D * D);
  cvt_bf16_kernel<<<blocks(D * D),  TPB, 0, stream>>>(wv, wvb, D * D);
  cvt_bf16_kernel<<<blocks(D * D),  TPB, 0, stream>>>(wg, wgb, D * D);
  cvt_bf16_kernel<<<blocks(D * D),  TPB, 0, stream>>>(wo, wob, D * D);

  // 2) projections: q,k,v,g = x @ W^T   (WMMA GEMM, 128x128 block tiles)
  dim3 ggrid(D / 128, MB / 128);
  gemm_bf16_wmma_kernel<<<ggrid, TPB, 0, stream>>>(xb, wqb, qf, MB, D, D);
  gemm_bf16_wmma_kernel<<<ggrid, TPB, 0, stream>>>(xb, wkb, kf, MB, D, D);
  gemm_bf16_wmma_kernel<<<ggrid, TPB, 0, stream>>>(xb, wvb, vf, MB, D, D);
  gemm_bf16_wmma_kernel<<<ggrid, TPB, 0, stream>>>(xb, wgb, gf, MB, D, D);

  // 3) RoPE(q,k) + bf16; V -> bf16 transposed [B,H,Dh,S]
  rope_bf16_kernel<<<blocks(B * S * H * (Dh / 2)), TPB, 0, stream>>>(qf, qbb, B, S, H, Dh);
  rope_bf16_kernel<<<blocks(B * S * H * (Dh / 2)), TPB, 0, stream>>>(kf, kbb, B, S, H, Dh);
  vT_bf16_kernel<<<blocks(B * S * H * Dh), TPB, 0, stream>>>(vf, vtb, B, S, H, Dh);

  // 4) streaming retention (flash-style, decay fused, no S x S materialization)
  dim3 rgrid(B * H, S / 128);
  retention_kernel<<<rgrid, TPB, 0, stream>>>(qbb, kbb, vtb, attf, B, S, H, Dh);

  // 5) SiLU gating fused with bf16 convert
  gate_silu_kernel<<<blocks(MB * D), TPB, 0, stream>>>(attf, gf, gatedb, MB * D);

  // 6) output projection: out = gated @ wo^T
  gemm_bf16_wmma_kernel<<<ggrid, TPB, 0, stream>>>(gatedb, wob, out, MB, D, D);

  (void)n_in; (void)out_size; (void)ws_size; (void)off;
}
